// SimpleLSTM_53377853555399
// MI455X (gfx1250) — matrix operand order repaired and checked by reading, not compiled, not run on hardware
//
#include <hip/hip_runtime.h>

// ---------------------------------------------------------------------------
// Types for CDNA5 WMMA (wave32, 16x16x32 bf16 -> f32)
// ---------------------------------------------------------------------------
typedef __bf16 bf16_t;
typedef __attribute__((ext_vector_type(16))) __bf16 v16bf;
typedef __attribute__((ext_vector_type(8)))  float  v8f;

union Frag32 { uint4 q[2]; v16bf v; unsigned short u[16]; };  // 32B: one A or B fragment per lane
union Pack8  { uint4 q;    unsigned short u[8]; };            // 16B: 8 bf16 per lane (D layout)
union Pack4  { unsigned long long q; unsigned short u[4]; };  // 8B: 4 bf16

__device__ __forceinline__ unsigned short f2bf(float f) {
    __bf16 h = (__bf16)f;                       // RNE fptrunc
    return __builtin_bit_cast(unsigned short, h);
}
__device__ __forceinline__ float bf2f(unsigned short u) {
    return (float)__builtin_bit_cast(__bf16, u);
}
__device__ __forceinline__ v8f vzero8() {
    v8f z;
    #pragma unroll
    for (int i = 0; i < 8; ++i) z[i] = 0.0f;
    return z;
}
__device__ __forceinline__ v8f wmma_bf16(v16bf a, v16bf b, v8f c) {
    // D = A(16x32) * B(32x16) + C, f32 accumulate
    return __builtin_amdgcn_wmma_f32_16x16x32_bf16(false, a, false, b, (short)0, c, false, false);
}
__device__ __forceinline__ float sigm(float x)   { return 1.0f / (1.0f + __expf(-x)); }
__device__ __forceinline__ float tanh_f(float x) { return 1.0f - 2.0f / (__expf(2.0f * x) + 1.0f); }

// Problem constants
#define BB   32
#define TT   4096
#define DIN  256
#define HH   256
#define G4H  1024

// ---------------------------------------------------------------------------
// Kernel 0: zero the h exchange buffer and the grid-barrier counter
// ---------------------------------------------------------------------------
__global__ void __launch_bounds__(256) init_state_kernel(unsigned short* h_buf, unsigned int* cnt) {
    for (int i = threadIdx.x; i < 2 * BB * HH; i += 256) h_buf[i] = 0;
    if (threadIdx.x == 0) cnt[0] = 0u;
}

// ---------------------------------------------------------------------------
// Kernel 1: xg = x @ Wx   (M = T*32 rows ordered rM = t*32+b, N = 1024, K = 256)
// Output written in WMMA D-fragment layout:
//   frag F = (t*2 + bhalf)*64 + ntile ; lane stores its 8 bf16 at F*256 + lane*8
// ---------------------------------------------------------------------------
__global__ void __launch_bounds__(256) gemm_xg_kernel(const float* __restrict__ x,
                                                      const float* __restrict__ wx,
                                                      unsigned short* __restrict__ xg) {
    __shared__ alignas(16) unsigned short a_sh[128 * 32];    // A tile, row-major bf16
    __shared__ alignas(16) unsigned short b_sh[4 * 32 * 16]; // B tile, fragment layout
    const int tid  = threadIdx.x;
    const int lane = tid & 31;
    const int w    = tid >> 5;
    const int wm   = w & 3, wn = w >> 2;     // 4x2 wave grid over 128x64 tile
    const int m0   = blockIdx.x * 128;
    const int n0   = blockIdx.y * 64;

    v8f acc[2][2];
    #pragma unroll
    for (int a = 0; a < 2; ++a)
        #pragma unroll
        for (int b = 0; b < 2; ++b) acc[a][b] = vzero8();

    for (int kt = 0; kt < 8; ++kt) {
        __syncthreads();
        // Stage A: gather rows rM = t*32+b from x[B,T,DIN]; b128 loads, packed b64 LDS stores
        #pragma unroll
        for (int p = 0; p < 4; ++p) {
            int id  = p * 256 + tid;        // 1024 float4
            int row = id >> 3;              // 8 float4 per 32-col row
            int c4  = (id & 7) * 4;
            int rM  = m0 + row;
            int t   = rM >> 5, bb2 = rM & 31;
            float4 v = *(const float4*)&x[((size_t)bb2 * TT + t) * DIN + kt * 32 + c4];
            Pack4 pk;
            pk.u[0] = f2bf(v.x); pk.u[1] = f2bf(v.y);
            pk.u[2] = f2bf(v.z); pk.u[3] = f2bf(v.w);
            *(unsigned long long*)&a_sh[row * 32 + c4] = pk.q;
        }
        // Stage B: Wx f32 -> bf16, swizzled to B-fragment layout (b128 loads)
        #pragma unroll
        for (int p = 0; p < 2; ++p) {
            int id = p * 256 + tid;         // 512 float4
            int k  = id >> 4;               // 16 float4 per 64-col k-row
            int n4 = (id & 15) * 4;
            float4 v = *(const float4*)&wx[(size_t)(kt * 32 + k) * G4H + n0 + n4];
            float vv[4] = {v.x, v.y, v.z, v.w};
            #pragma unroll
            for (int j = 0; j < 4; ++j) {
                int n  = n4 + j;
                int lb = (n & 15) + ((k >> 4) << 4);
                b_sh[((n >> 4) * 32 + lb) * 16 + (k & 15)] = f2bf(vv[j]);
            }
        }
        __syncthreads();
        #pragma unroll
        for (int mt = 0; mt < 2; ++mt) {
            Frag32 a;
            int row = wm * 32 + mt * 16 + (lane & 15);
            int kb  = (lane >> 4) * 16;
            const uint4* pa = (const uint4*)&a_sh[row * 32 + kb];
            a.q[0] = pa[0]; a.q[1] = pa[1];
            #pragma unroll
            for (int ln = 0; ln < 2; ++ln) {
                Frag32 bf_;
                const uint4* pb = (const uint4*)&b_sh[((wn * 2 + ln) * 32 + lane) * 16];
                bf_.q[0] = pb[0]; bf_.q[1] = pb[1];
                acc[mt][ln] = wmma_bf16(a.v, bf_.v, acc[mt][ln]);
            }
        }
    }
    // Epilogue: one b128 store per (mt,ln) per lane, fragment layout
    #pragma unroll
    for (int mt = 0; mt < 2; ++mt) {
        int rowbase = m0 + wm * 32 + mt * 16;
        int t = rowbase >> 5;
        int mhalf = (rowbase >> 4) & 1;
        #pragma unroll
        for (int ln = 0; ln < 2; ++ln) {
            int ntg = (n0 + wn * 32 + ln * 16) >> 4;
            size_t F = ((size_t)t * 2 + mhalf) * 64 + ntg;
            Pack8 pk;
            #pragma unroll
            for (int j = 0; j < 8; ++j) pk.u[j] = f2bf(acc[mt][ln][j]);
            *(uint4*)(xg + F * 256 + lane * 8) = pk.q;
        }
    }
}

// ---------------------------------------------------------------------------
// Kernel 2: persistent LSTM recurrence. 8 WGs x 128 threads (4 waves).
// WG g owns h columns [32g, 32g+32) and gate columns {q*256 + 32g .. +32}.
// Wh slice resident in LDS (K-tiles 0..6, B-fragment layout) + registers (K-tile 7).
// One device barrier per step; h exchanged via L2-resident double buffer.
// Next step's xg fragments prefetched (global_prefetch) behind the WMMA chain.
// ---------------------------------------------------------------------------
__global__ void __launch_bounds__(128) lstm_rec_kernel(const unsigned short* __restrict__ xg,
                                                       const float* __restrict__ wh,
                                                       const float* __restrict__ bias,
                                                       unsigned short* __restrict__ h_buf,
                                                       unsigned int* __restrict__ cnt,
                                                       unsigned short* __restrict__ h_st) {
    __shared__ alignas(16) unsigned short wh_sh[8 * 7 * 32 * 16]; // 56KB: [ln][kt<7][lane][16]
    __shared__ alignas(16) unsigned short hstage[32 * 32];        // 2KB: local h tile row-major
    const int tid  = threadIdx.x;
    const int lane = tid & 31;
    const int w    = tid >> 5;
    const int mt   = w & 1;   // batch half (rows 0-15 / 16-31)
    const int hn   = w >> 1;  // h-column 16-tile within WG slice
    const int g    = blockIdx.x;

    // Stage Wh K-tiles 0..6 into LDS in B-fragment layout (one time)
    for (int idx = tid; idx < 8 * 7 * 512; idx += 128) {
        int klo = idx & 15;
        int ln2 = (idx >> 4) & 31;
        int rest = idx >> 9;
        int kt = rest % 7;
        int ln = rest / 7;
        int q = ln >> 1, h2 = ln & 1;
        int k   = kt * 32 + ((ln2 >> 4) << 4) + klo;
        int col = q * 256 + g * 32 + h2 * 16 + (ln2 & 15);
        wh_sh[idx] = f2bf(wh[(size_t)k * G4H + col]);
    }
    // K-tile 7 B-fragments in registers + per-lane gate bias
    Frag32 breg[4];
    float bval[4];
    #pragma unroll
    for (int q = 0; q < 4; ++q) {
        int col = q * 256 + g * 32 + hn * 16 + (lane & 15);
        int kb  = 7 * 32 + ((lane >> 4) << 4);
        #pragma unroll
        for (int j = 0; j < 16; ++j) breg[q].u[j] = f2bf(wh[(size_t)(kb + j) * G4H + col]);
        bval[q] = bias[col];
    }
    __syncthreads();

    v8f c = vzero8();
    const int arow = mt * 16 + (lane & 15);
    const int kb8  = (lane >> 4) * 16;

    for (int t = 0; t < TT; ++t) {
        const int buf = t & 1;
        v8f acc[4];
        #pragma unroll
        for (int q = 0; q < 4; ++q) acc[q] = vzero8();

        // Prefetch next step's xg fragments into cache behind this step's WMMAs
        if (t + 1 < TT) {
            size_t Fn = ((size_t)(t + 1) * 2 + mt) * 64;
            #pragma unroll
            for (int q = 0; q < 4; ++q)
                __builtin_prefetch((const void*)(xg + (Fn + q * 16 + g * 2 + hn) * 256 + lane * 8), 0, 3);
        }

        #pragma unroll
        for (int kt = 0; kt < 8; ++kt) {
            Frag32 a;  // A fragment of h(t) straight from L2-resident buffer
            const uint4* pa = (const uint4*)&h_buf[((size_t)buf * 32 + arow) * HH + kt * 32 + kb8];
            a.q[0] = pa[0]; a.q[1] = pa[1];
            #pragma unroll
            for (int q = 0; q < 4; ++q) {
                Frag32 bf_;
                if (kt < 7) {
                    const uint4* pb = (const uint4*)&wh_sh[(((q * 2 + hn) * 7 + kt) * 32 + lane) * 16];
                    bf_.q[0] = pb[0]; bf_.q[1] = pb[1];
                } else {
                    bf_ = breg[q];
                }
                acc[q] = wmma_bf16(a.v, bf_.v, acc[q]);
            }
        }
        // gates = h@Wh + xg(frag layout, 1 b128/gate) + bias; then LSTM cell
        size_t Fb = ((size_t)t * 2 + mt) * 64;
        float gv[4][8];
        #pragma unroll
        for (int q = 0; q < 4; ++q) {
            size_t F = Fb + q * 16 + g * 2 + hn;
            Pack8 xk; xk.q = *(const uint4*)(xg + F * 256 + lane * 8);
            #pragma unroll
            for (int j = 0; j < 8; ++j) gv[q][j] = acc[q][j] + bf2f(xk.u[j]) + bval[q];
        }
        #pragma unroll
        for (int j = 0; j < 8; ++j) {
            float ci = sigm(gv[1][j]) * c[j] + sigm(gv[0][j]) * tanh_f(gv[2][j]);
            c[j] = ci;
            hstage[(mt * 16 + j + 8 * (lane >> 4)) * 32 + hn * 16 + (lane & 15)] =
                f2bf(sigm(gv[3][j]) * tanh_f(ci));
        }
        __syncthreads();
        {   // coalesced b128 publish of this WG's 32x32 h slice
            int r2 = tid >> 2;
            int c8 = (tid & 3) * 8;
            uint4 v = *(const uint4*)&hstage[r2 * 32 + c8];
            *(uint4*)&h_buf[((size_t)(buf ^ 1) * 32 + r2) * HH + g * 32 + c8] = v;
            *(uint4*)&h_st[((size_t)t * 32 + r2) * HH + g * 32 + c8] = v;
        }
        __threadfence();       // release h(t+1)
        __syncthreads();
        if (tid == 0) {
            atomicAdd(cnt, 1u);
            unsigned target = 8u * (unsigned)(t + 1);
            while (__hip_atomic_load(cnt, __ATOMIC_RELAXED, __HIP_MEMORY_SCOPE_AGENT) < target)
                __builtin_amdgcn_s_sleep(2);
        }
        __syncthreads();
        __threadfence();       // acquire: invalidate stale h lines for all lanes
    }
}

// ---------------------------------------------------------------------------
// Kernel 3: out = h_stacked @ Wo + bo   (A bf16 row-major [T*32,256], out f32 [B,T,256])
// ---------------------------------------------------------------------------
__global__ void __launch_bounds__(256) gemm_out_kernel(const unsigned short* __restrict__ h_st,
                                                       const float* __restrict__ wo,
                                                       const float* __restrict__ bo,
                                                       float* __restrict__ out) {
    __shared__ alignas(16) unsigned short a_sh[128 * 32];
    __shared__ alignas(16) unsigned short b_sh[4 * 32 * 16];
    __shared__ alignas(16) float o_sh[128 * 64];
    const int tid  = threadIdx.x;
    const int lane = tid & 31;
    const int w    = tid >> 5;
    const int wm   = w & 3, wn = w >> 2;
    const int m0   = blockIdx.x * 128;
    const int n0   = blockIdx.y * 64;

    float bias[2];
    #pragma unroll
    for (int ln = 0; ln < 2; ++ln) bias[ln] = bo[n0 + wn * 32 + ln * 16 + (lane & 15)];

    v8f acc[2][2];
    #pragma unroll
    for (int a = 0; a < 2; ++a)
        #pragma unroll
        for (int b = 0; b < 2; ++b) acc[a][b] = vzero8();

    for (int kt = 0; kt < 8; ++kt) {
        __syncthreads();
        // Stage A: contiguous bf16 rows, b64 copies
        #pragma unroll
        for (int p = 0; p < 4; ++p) {
            int id  = p * 256 + tid;          // 1024 x 8B
            int row = id >> 3;
            int c4  = (id & 7) * 4;
            *(unsigned long long*)&a_sh[row * 32 + c4] =
                *(const unsigned long long*)&h_st[(size_t)(m0 + row) * HH + kt * 32 + c4];
        }
        // Stage B: Wo f32 -> bf16 fragment layout (b128 loads)
        #pragma unroll
        for (int p = 0; p < 2; ++p) {
            int id = p * 256 + tid;           // 512 float4
            int k  = id >> 4;
            int n4 = (id & 15) * 4;
            float4 v = *(const float4*)&wo[(size_t)(kt * 32 + k) * 256 + n0 + n4];
            float vv[4] = {v.x, v.y, v.z, v.w};
            #pragma unroll
            for (int j = 0; j < 4; ++j) {
                int n  = n4 + j;
                int lb = (n & 15) + ((k >> 4) << 4);
                b_sh[((n >> 4) * 32 + lb) * 16 + (k & 15)] = f2bf(vv[j]);
            }
        }
        __syncthreads();
        #pragma unroll
        for (int mt = 0; mt < 2; ++mt) {
            Frag32 a;
            int row = wm * 32 + mt * 16 + (lane & 15);
            int kb  = (lane >> 4) * 16;
            const uint4* pa = (const uint4*)&a_sh[row * 32 + kb];
            a.q[0] = pa[0]; a.q[1] = pa[1];
            #pragma unroll
            for (int ln = 0; ln < 2; ++ln) {
                Frag32 bf_;
                const uint4* pb = (const uint4*)&b_sh[((wn * 2 + ln) * 32 + lane) * 16];
                bf_.q[0] = pb[0]; bf_.q[1] = pb[1];
                acc[mt][ln] = wmma_bf16(a.v, bf_.v, acc[mt][ln]);
            }
        }
    }
    __syncthreads();
    // Bounce through LDS for coalesced f32x4 stores into [B,T,D] output
    #pragma unroll
    for (int mt = 0; mt < 2; ++mt)
        #pragma unroll
        for (int ln = 0; ln < 2; ++ln)
            #pragma unroll
            for (int j = 0; j < 8; ++j)
                o_sh[(wm * 32 + mt * 16 + j + 8 * (lane >> 4)) * 64 +
                     wn * 32 + ln * 16 + (lane & 15)] = acc[mt][ln][j] + bias[ln];
    __syncthreads();
    #pragma unroll
    for (int p = 0; p < 8; ++p) {
        int id  = p * 256 + tid;          // 2048 x 16B
        int row = id >> 4;
        int q4  = (id & 15) * 4;
        int rM  = m0 + row;
        int t = rM >> 5, bb = rM & 31;
        *(uint4*)(out + ((size_t)bb * TT + t) * 256 + n0 + q4) = *(const uint4*)&o_sh[row * 64 + q4];
    }
}

// ---------------------------------------------------------------------------
extern "C" void kernel_launch(void* const* d_in, const int* in_sizes, int n_in,
                              void* d_out, int out_size, void* d_ws, size_t ws_size,
                              hipStream_t stream) {
    const float* x  = (const float*)d_in[0];
    const float* Wx = (const float*)d_in[1];
    const float* Wh = (const float*)d_in[2];
    const float* b  = (const float*)d_in[3];
    const float* Wo = (const float*)d_in[4];
    const float* bo = (const float*)d_in[5];
    float* out = (float*)d_out;

    char* ws = (char*)d_ws;
    const size_t xg_bytes  = (size_t)TT * BB * G4H * 2;   // 256 MiB, bf16 fragment layout
    const size_t hst_bytes = (size_t)TT * BB * HH * 2;    // 64 MiB, bf16 row-major [T*32,H]
    unsigned short* xg    = (unsigned short*)ws;
    unsigned short* h_st  = (unsigned short*)(ws + xg_bytes);
    unsigned short* h_buf = (unsigned short*)(ws + xg_bytes + hst_bytes);
    unsigned int*   cnt   = (unsigned int*)(ws + xg_bytes + hst_bytes + (size_t)2 * BB * HH * 2);

    init_state_kernel<<<1, 256, 0, stream>>>(h_buf, cnt);
    gemm_xg_kernel<<<dim3((TT * BB) / 128, G4H / 64), 256, 0, stream>>>(x, Wx, xg);
    lstm_rec_kernel<<<8, 128, 0, stream>>>(xg, Wh, b, h_buf, cnt, h_st);
    gemm_out_kernel<<<dim3((TT * BB) / 128, 256 / 64), 256, 0, stream>>>(h_st, Wo, bo, out);
}
